// GATModel_3461743641205
// MI455X (gfx1250) — matrix operand order repaired and checked by reading, not compiled, not run on hardware
//
#include <hip/hip_runtime.h>

#define NEG_SLOPE 0.2f

typedef __attribute__((ext_vector_type(16))) _Float16 v16h;
typedef __attribute__((ext_vector_type(8)))  _Float16 v8h;
typedef __attribute__((ext_vector_type(8)))  float    v8f;

// ---- ordered-uint encoding so unsigned atomicMax == float max (monotone) ----
__device__ __forceinline__ unsigned f2o(float f){
  unsigned u = __float_as_uint(f);
  return (u & 0x80000000u) ? ~u : (u | 0x80000000u);
}
__device__ __forceinline__ float o2f(unsigned u){
  return (u & 0x80000000u) ? __uint_as_float(u & 0x7FFFFFFFu) : __uint_as_float(~u);
}

__global__ void k_f32_to_f16(const float* __restrict__ in, _Float16* __restrict__ out, int n){
  int i = blockIdx.x*blockDim.x + threadIdx.x;
  if (i < n) out[i] = (_Float16)in[i];
}

// W[K][NC] row-major (f32) -> WT[NC][K] (f16)
__global__ void k_transpose_f16(const float* __restrict__ W, _Float16* __restrict__ WT, int K, int NC){
  int i = blockIdx.x*blockDim.x + threadIdx.x;
  if (i >= K*NC) return;
  int k = i / NC, j = i - k*NC;
  WT[(size_t)j*K + k] = (_Float16)W[i];
}

// C[M,NC] = A[M,K] x B[K,NC]; A row-major f16, B passed as BT[NC][K] f16, C f32.
// One wave per 16x16 output tile; v_wmma_f32_16x16x32_f16, fp32 accumulate.
__global__ __launch_bounds__(128) void k_wmma_gemm(const _Float16* __restrict__ A,
                                                   const _Float16* __restrict__ BT,
                                                   float* __restrict__ C,
                                                   int mTiles, int K, int NC){
  int wave  = threadIdx.x >> 5;
  int lane  = threadIdx.x & 31;
  int mTile = blockIdx.x*4 + wave;
  if (mTile >= mTiles) return;           // wave-uniform exit: EXEC all-ones at WMMA
  int nTile = blockIdx.y;
  int half  = lane >> 4;                 // 0: lanes 0-15, 1: lanes 16-31
  int l15   = lane & 15;
  const _Float16* arow = A  + (size_t)(mTile*16 + l15) * K;   // A row M = l15
  const _Float16* bcol = BT + (size_t)(nTile*16 + l15) * K;   // B column N = l15
  v8f acc = {0.f,0.f,0.f,0.f,0.f,0.f,0.f,0.f};
  for (int k0 = 0; k0 < K; k0 += 32){
    // A 16x32 f16 layout: lane<16 holds K = 0..7,16..23 ; lane>=16 holds K = 8..15,24..31
    v8h alo = *(const v8h*)(arow + k0 + half*8);
    v8h ahi = *(const v8h*)(arow + k0 + 16 + half*8);
    v16h a;
    #pragma unroll
    for (int j = 0; j < 8; ++j){ a[j] = alo[j]; a[j+8] = ahi[j]; }
    // B 32x16 f16 layout: lane<16 holds K = 0..7,16..23 ; lane>=16 holds K = 8..15,24..31 of col l15
    v8h blo = *(const v8h*)(bcol + k0 + half*8);
    v8h bhi = *(const v8h*)(bcol + k0 + 16 + half*8);
    v16h b;
    #pragma unroll
    for (int j = 0; j < 8; ++j){ b[j] = blo[j]; b[j+8] = bhi[j]; }
    acc = __builtin_amdgcn_wmma_f32_16x16x32_f16(false, a, false, b, (short)0, acc, false, false);
  }
  // D layout: VGPR r -> row (r + 8*half), col l15
  float* crow = C + (size_t)(mTile*16 + half*8) * NC + (size_t)nTile*16 + l15;
  #pragma unroll
  for (int r = 0; r < 8; ++r) crow[(size_t)r*NC] = acc[r];
}

// per (node, head): alpha_src/alpha_dst channel dots
__global__ void k_alpha(const float* __restrict__ h, const float* __restrict__ a_s,
                        const float* __restrict__ a_d, float* __restrict__ as,
                        float* __restrict__ ad, int N, int H, int C){
  int i = blockIdx.x*blockDim.x + threadIdx.x;
  if (i >= N*H) return;
  int hh = i % H;
  const float* hp = h + (size_t)i * C;    // h is [N,H,C], i = n*H+hh
  float s = 0.f, d = 0.f;
  for (int c = 0; c < C; ++c){ s += hp[c]*a_s[hh*C+c]; d += hp[c]*a_d[hh*C+c]; }
  as[i] = s; ad[i] = d;
}

__device__ __forceinline__ void edge_sd(const long long* ei, int E, int e, int& s, int& d){
  if (e < E){ s = (int)ei[e]; d = (int)ei[(size_t)E + e]; }
  else      { s = d = e - E; }            // implicit self-loop edges
}

__global__ void k_edge_max(const long long* __restrict__ ei, int E, int N,
                           const float* __restrict__ as, const float* __restrict__ ad,
                           unsigned* __restrict__ m, int H){
  int t = blockIdx.x*blockDim.x + threadIdx.x;
  if (t >= (E+N)*H) return;
  int e = t / H, h = t - e*H;
  int s, d; edge_sd(ei, E, e, s, d);
  float v = as[(size_t)s*H + h] + ad[(size_t)d*H + h];
  v = v > 0.f ? v : NEG_SLOPE*v;
  atomicMax(&m[(size_t)d*H + h], f2o(v));
}

__global__ void k_edge_expsum(const long long* __restrict__ ei, int E, int N,
                              const float* __restrict__ as, const float* __restrict__ ad,
                              const unsigned* __restrict__ m, float* __restrict__ den, int H){
  int t = blockIdx.x*blockDim.x + threadIdx.x;
  if (t >= (E+N)*H) return;
  int e = t / H, h = t - e*H;
  int s, d; edge_sd(ei, E, e, s, d);
  float v = as[(size_t)s*H + h] + ad[(size_t)d*H + h];
  v = v > 0.f ? v : NEG_SLOPE*v;
  atomicAdd(&den[(size_t)d*H + h], expf(v - o2f(m[(size_t)d*H + h])));
}

// one wave per edge: amortize src/dst/alpha loads over all H*C channels
__global__ void k_edge_agg(const long long* __restrict__ ei, int E, int N,
                           const float* __restrict__ as, const float* __restrict__ ad,
                           const unsigned* __restrict__ m, const float* __restrict__ den,
                           const float* __restrict__ feat, float* __restrict__ out,
                           int H, int C){
  int gt = blockIdx.x*blockDim.x + threadIdx.x;
  int w = gt >> 5, lane = gt & 31;
  if (w >= E+N) return;
  int s, d; edge_sd(ei, E, w, s, d);
  for (int h = 0; h < H; ++h){
    float v = as[(size_t)s*H + h] + ad[(size_t)d*H + h];
    v = v > 0.f ? v : NEG_SLOPE*v;
    float alpha = expf(v - o2f(m[(size_t)d*H + h])) / den[(size_t)d*H + h];
    const float* fs = feat + ((size_t)s*H + h)*C;
    float*       od = out  + ((size_t)d*H + h)*C;
    for (int c = lane; c < C; c += 32)
      atomicAdd(&od[c], alpha * fs[c]);
  }
}

// layer-1 epilogue: relu(agg + b1) -> f16 for GEMM2
__global__ void k_relu_bias_f16(const float* __restrict__ agg, const float* __restrict__ b,
                                _Float16* __restrict__ out, int total, int F){
  int i = blockIdx.x*blockDim.x + threadIdx.x;
  if (i >= total) return;
  float v = agg[i] + b[i % F];
  out[i] = (_Float16)(v > 0.f ? v : 0.f);
}

__global__ void k_add_bias(float* __restrict__ out, const float* __restrict__ b, int total, int C){
  int i = blockIdx.x*blockDim.x + threadIdx.x;
  if (i >= total) return;
  out[i] += b[i % C];
}

static inline int ceil_div(long long a, int b){ return (int)((a + b - 1) / b); }

extern "C" void kernel_launch(void* const* d_in, const int* in_sizes, int n_in,
                              void* d_out, int out_size, void* d_ws, size_t ws_size,
                              hipStream_t stream) {
  const float*     x      = (const float*)d_in[0];
  const long long* ei     = (const long long*)d_in[1];   // [2,E] int64
  const float*     W1     = (const float*)d_in[3];
  const float*     a_src1 = (const float*)d_in[4];
  const float*     a_dst1 = (const float*)d_in[5];
  const float*     b1     = (const float*)d_in[6];
  const float*     W2     = (const float*)d_in[7];
  const float*     a_src2 = (const float*)d_in[8];
  const float*     a_dst2 = (const float*)d_in[9];
  const float*     b2     = (const float*)d_in[10];

  const int IN_C = 128, HID = 32, H1 = 4, OUT_C = 64;
  const int F1 = H1 * HID;                 // 128
  int N  = in_sizes[0] / IN_C;             // 50000
  int E  = in_sizes[1] / 2;                // 800000
  int E2 = E + N;                          // edges incl. self-loops
  float* out = (float*)d_out;              // [N, OUT_C]

  // ---- workspace bump allocation (stream-ordered reuse via aliasing) ----
  char* ws = (char*)d_ws; size_t off = 0;
  auto alloc = [&](size_t bytes)->void* {
    void* p = ws + off; off = (off + bytes + 255) & ~(size_t)255; return p;
  };
  _Float16* xh   = (_Float16*)alloc((size_t)N*IN_C*sizeof(_Float16));
  _Float16* W1T  = (_Float16*)alloc((size_t)F1*IN_C*sizeof(_Float16));
  _Float16* W2T  = (_Float16*)alloc((size_t)OUT_C*F1*sizeof(_Float16));
  float*    h1   = (float*)alloc((size_t)N*F1*sizeof(float));
  float*    agg1 = (float*)alloc((size_t)N*F1*sizeof(float));
  float*    as1  = (float*)alloc((size_t)N*H1*sizeof(float));
  float*    ad1  = (float*)alloc((size_t)N*H1*sizeof(float));
  unsigned* m1   = (unsigned*)alloc((size_t)N*H1*sizeof(unsigned));
  float*    den1 = (float*)alloc((size_t)N*H1*sizeof(float));
  float*    as2  = (float*)alloc((size_t)N*sizeof(float));
  float*    ad2  = (float*)alloc((size_t)N*sizeof(float));
  unsigned* m2   = (unsigned*)alloc((size_t)N*sizeof(unsigned));
  float*    den2 = (float*)alloc((size_t)N*sizeof(float));
  _Float16* hrh  = xh;     // alias: x(f16) dead after GEMM1
  float*    h2   = agg1;   // alias: agg1 dead after relu epilogue

  const int B = 256;
  int mTiles = (N + 15) / 16;

  // ---- layer 1 ----
  k_f32_to_f16<<<ceil_div((long long)N*IN_C, B), B, 0, stream>>>(x, xh, N*IN_C);
  k_transpose_f16<<<ceil_div((long long)IN_C*F1, B), B, 0, stream>>>(W1, W1T, IN_C, F1);
  {
    dim3 g(ceil_div(mTiles, 4), F1/16);
    k_wmma_gemm<<<g, 128, 0, stream>>>(xh, W1T, h1, mTiles, IN_C, F1);
  }
  k_alpha<<<ceil_div((long long)N*H1, B), B, 0, stream>>>(h1, a_src1, a_dst1, as1, ad1, N, H1, HID);

  hipMemsetAsync(m1,   0, (size_t)N*H1*sizeof(unsigned), stream);  // f2o-min
  hipMemsetAsync(den1, 0, (size_t)N*H1*sizeof(float),    stream);
  hipMemsetAsync(agg1, 0, (size_t)N*F1*sizeof(float),    stream);

  k_edge_max   <<<ceil_div((long long)E2*H1, B), B, 0, stream>>>(ei, E, N, as1, ad1, m1, H1);
  k_edge_expsum<<<ceil_div((long long)E2*H1, B), B, 0, stream>>>(ei, E, N, as1, ad1, m1, den1, H1);
  k_edge_agg   <<<ceil_div((long long)E2*32, B), B, 0, stream>>>(ei, E, N, as1, ad1, m1, den1,
                                                                 h1, agg1, H1, HID);
  k_relu_bias_f16<<<ceil_div((long long)N*F1, B), B, 0, stream>>>(agg1, b1, hrh, N*F1, F1);

  // ---- layer 2 (H=1, C=64, mean over 1 head == identity) ----
  k_transpose_f16<<<ceil_div((long long)F1*OUT_C, B), B, 0, stream>>>(W2, W2T, F1, OUT_C);
  {
    dim3 g(ceil_div(mTiles, 4), OUT_C/16);
    k_wmma_gemm<<<g, 128, 0, stream>>>(hrh, W2T, h2, mTiles, F1, OUT_C);
  }
  k_alpha<<<ceil_div((long long)N, B), B, 0, stream>>>(h2, a_src2, a_dst2, as2, ad2, N, 1, OUT_C);

  hipMemsetAsync(m2,   0, (size_t)N*sizeof(unsigned), stream);
  hipMemsetAsync(den2, 0, (size_t)N*sizeof(float),    stream);
  hipMemsetAsync(out,  0, (size_t)N*OUT_C*sizeof(float), stream);

  k_edge_max   <<<ceil_div((long long)E2, B), B, 0, stream>>>(ei, E, N, as2, ad2, m2, 1);
  k_edge_expsum<<<ceil_div((long long)E2, B), B, 0, stream>>>(ei, E, N, as2, ad2, m2, den2, 1);
  k_edge_agg   <<<ceil_div((long long)E2*32, B), B, 0, stream>>>(ei, E, N, as2, ad2, m2, den2,
                                                                 h2, out, 1, OUT_C);
  k_add_bias<<<ceil_div((long long)N*OUT_C, B), B, 0, stream>>>(out, b2, N*OUT_C, OUT_C);
}
